// CachedMultiHeadAttention_90245852823566
// MI455X (gfx1250) — matrix operand order repaired and checked by reading, not compiled, not run on hardware
//
#include <hip/hip_runtime.h>
#include <hip/hip_bf16.h>
#include <math.h>

#define B_  32
#define S_  2048
#define D_  2048
#define H_  16
#define HD_ 128

typedef __attribute__((ext_vector_type(16))) _Float16 v16h;
typedef __attribute__((ext_vector_type(8)))  float    v8f;
typedef __attribute__((ext_vector_type(4)))  float    v4f;

// ---------------------------------------------------------------------------
// GEMM: Out[m,n] = sum_k X[m,k] * W[n,k] + bias[n];  M = 32, N = K = 2048.
// One wave computes a 32x16 output strip (two 16x16 WMMA tiles sharing the
// same B tile, so each weight matrix is streamed from memory exactly once).
// Block = 256 threads = 8 waves -> 8 consecutive N tiles. Grid = (16, 1).
// ---------------------------------------------------------------------------
__global__ __launch_bounds__(256)
void gemm_xwT_wmma(const float* __restrict__ X, const float* __restrict__ W,
                   const float* __restrict__ bias, float* __restrict__ Out)
{
    const int lane   = threadIdx.x & 31;
    const int wave   = threadIdx.x >> 5;
    const int n_tile = (blockIdx.x * 8 + wave) * 16;

    // A layout (16-bit A 16x32): lane L -> row m = L&15; element i ->
    // k = 16*(i/8) + (L>=16 ? 8 : 0) + (i%8)
    const int    kb    = (lane & 16) ? 8 : 0;
    const float* xrow0 = X + (size_t)(lane & 15) * D_;        // m-tile 0..15
    const float* xrow1 = X + (size_t)(16 + (lane & 15)) * D_; // m-tile 16..31

    v8f acc0 = {};
    v8f acc1 = {};
    #pragma unroll 2
    for (int k0 = 0; k0 < D_; k0 += 32) {
        v4f a0 = *(const v4f*)(xrow0 + k0 + kb);
        v4f a1 = *(const v4f*)(xrow0 + k0 + kb + 4);
        v4f a2 = *(const v4f*)(xrow0 + k0 + kb + 16);
        v4f a3 = *(const v4f*)(xrow0 + k0 + kb + 20);
        v4f c0 = *(const v4f*)(xrow1 + k0 + kb);
        v4f c1 = *(const v4f*)(xrow1 + k0 + kb + 4);
        v4f c2 = *(const v4f*)(xrow1 + k0 + kb + 16);
        v4f c3 = *(const v4f*)(xrow1 + k0 + kb + 20);

        v16h A0, A1;
        #pragma unroll
        for (int j = 0; j < 4; ++j) {
            A0[j]      = (_Float16)a0[j];
            A0[4 + j]  = (_Float16)a1[j];
            A0[8 + j]  = (_Float16)a2[j];
            A0[12 + j] = (_Float16)a3[j];
            A1[j]      = (_Float16)c0[j];
            A1[4 + j]  = (_Float16)c1[j];
            A1[8 + j]  = (_Float16)c2[j];
            A1[12 + j] = (_Float16)c3[j];
        }

        // B layout (16-bit B 32x16): lane L -> col n = L&15; element i ->
        // k = 16*(i/8) + (L>=16 ? 8 : 0) + (i%8);  B[k][n] = W[n_tile+n][k0+k]
        v16h Bv;
        #pragma unroll
        for (int i = 0; i < 16; ++i)
            Bv[i] = (_Float16) W[(size_t)(n_tile + (lane & 15)) * D_ + k0 + kb + i + (i >= 8 ? 8 : 0)];

        acc0 = __builtin_amdgcn_wmma_f32_16x16x32_f16(
                   false, A0, false, Bv, (short)0, acc0, false, false);
        acc1 = __builtin_amdgcn_wmma_f32_16x16x32_f16(
                   false, A1, false, Bv, (short)0, acc1, false, false);
    }

    // D layout: lane L -> n = L&15; VGPR v -> m = v + 8*(L/16)
    const int   n  = n_tile + (lane & 15);
    const float bv = bias[n];
    const int   mb = (lane >> 4) * 8;
    #pragma unroll
    for (int v = 0; v < 8; ++v) {
        Out[(size_t)(mb + v) * D_ + n]      = acc0[v] + bv;
        Out[(size_t)(16 + mb + v) * D_ + n] = acc1[v] + bv;
    }
}

// ---------------------------------------------------------------------------
// Decode attention for one (b,h) per 256-thread block.
// Row `pos` of the caches is replaced by Knew/Vnew (inputs never mutated).
// K/V cache loads are non-temporal: 1.07 GB streamed once must not evict the
// L2-resident weight matrices (67 MB, reused every graph replay).
// ---------------------------------------------------------------------------
__global__ __launch_bounds__(256)
void attn_decode(const float* __restrict__ Q, const float* __restrict__ Knew,
                 const float* __restrict__ Vnew,
                 const float* __restrict__ kcache, const float* __restrict__ vcache,
                 const int* __restrict__ posp, float* __restrict__ Out)
{
    __shared__ float sc[S_];
    __shared__ float red[256];
    __shared__ __align__(16) float obuf[8][HD_];

    const int tid  = threadIdx.x;
    const int lane = tid & 31;
    const int wave = tid >> 5;
    const int b    = blockIdx.x / H_;
    const int h    = blockIdx.x % H_;
    const int pos  = *posp;
    const int nrow = pos + 1;

    const float* qptr = Q + (size_t)b * D_ + h * HD_;
    const v4f    q4   = ((const v4f*)qptr)[lane];

    const float* kbase = kcache + (size_t)(b * H_ + h) * S_ * HD_;
    const float* knrow = Knew + (size_t)b * D_ + h * HD_;
    const float  scale = 0.08838834764831845f;   // 1/sqrt(128)

    // scores: one K row per wave iteration, v4f per lane, wave reduction
    for (int s = wave; s < nrow; s += 8) {
        const float* kr = (s == pos) ? knrow : (kbase + (size_t)s * HD_);
        const v4f k4 = __builtin_nontemporal_load((const v4f*)kr + lane);
        float p = q4.x * k4.x + q4.y * k4.y + q4.z * k4.z + q4.w * k4.w;
        #pragma unroll
        for (int off = 16; off >= 1; off >>= 1) p += __shfl_xor(p, off, 32);
        if (lane == 0) sc[s] = p * scale;
    }
    __syncthreads();

    // softmax max
    float lmax = -3.0e38f;
    for (int s = tid; s < nrow; s += 256) lmax = fmaxf(lmax, sc[s]);
    red[tid] = lmax;
    __syncthreads();
    for (int w = 128; w > 0; w >>= 1) {
        if (tid < w) red[tid] = fmaxf(red[tid], red[tid + w]);
        __syncthreads();
    }
    const float mx = red[0];
    __syncthreads();

    // exp + sum
    float lsum = 0.f;
    for (int s = tid; s < nrow; s += 256) {
        const float e = __expf(sc[s] - mx);
        sc[s] = e;
        lsum += e;
    }
    red[tid] = lsum;
    __syncthreads();
    for (int w = 128; w > 0; w >>= 1) {
        if (tid < w) red[tid] += red[tid + w];
        __syncthreads();
    }
    const float inv = 1.0f / red[0];

    // weighted V accumulation (non-temporal stream)
    const float* vbase = vcache + (size_t)(b * H_ + h) * S_ * HD_;
    const float* vnrow = Vnew + (size_t)b * D_ + h * HD_;
    v4f acc = {0.f, 0.f, 0.f, 0.f};
    for (int s = wave; s < nrow; s += 8) {
        const float* vr = (s == pos) ? vnrow : (vbase + (size_t)s * HD_);
        const v4f v4 = __builtin_nontemporal_load((const v4f*)vr + lane);
        acc += sc[s] * v4;
    }
    *(v4f*)(&obuf[wave][4 * lane]) = acc;
    __syncthreads();

    if (tid < HD_) {
        float o = 0.f;
        #pragma unroll
        for (int w = 0; w < 8; ++w) o += obuf[w][tid];
        Out[(size_t)b * D_ + h * HD_ + tid] = o * inv;
    }
}

// ---------------------------------------------------------------------------
extern "C" void kernel_launch(void* const* d_in, const int* in_sizes, int n_in,
                              void* d_out, int out_size, void* d_ws, size_t ws_size,
                              hipStream_t stream)
{
    const float* query  = (const float*)d_in[0];
    const float* key    = (const float*)d_in[1];
    const float* value  = (const float*)d_in[2];
    const float* kcache = (const float*)d_in[3];
    const float* vcache = (const float*)d_in[4];
    const float* Wq     = (const float*)d_in[5];
    const float* bq     = (const float*)d_in[6];
    const float* Wk     = (const float*)d_in[7];
    const float* bk     = (const float*)d_in[8];
    const float* Wv     = (const float*)d_in[9];
    const float* bv     = (const float*)d_in[10];
    const float* Wo     = (const float*)d_in[11];
    const float* bo     = (const float*)d_in[12];
    const int*   pos    = (const int*)d_in[13];

    float* Qws = (float*)d_ws;            // 32*2048 f32
    float* Kws = Qws + B_ * D_;           // new K rows
    float* Vws = Kws + B_ * D_;           // new V rows
    float* Ows = Vws + B_ * D_;           // attention output (pre-projection)

    const dim3 gb(256);
    const dim3 gg(D_ / 16 / 8, 1);        // (16, 1): each wave does both M tiles

    gemm_xwT_wmma<<<gg, gb, 0, stream>>>(query, Wq, bq, Qws);
    gemm_xwT_wmma<<<gg, gb, 0, stream>>>(key,   Wk, bk, Kws);
    gemm_xwT_wmma<<<gg, gb, 0, stream>>>(value, Wv, bv, Vws);
    attn_decode<<<dim3(B_ * H_), dim3(256), 0, stream>>>(Qws, Kws, Vws,
                                                         kcache, vcache, pos, Ows);
    gemm_xwT_wmma<<<gg, gb, 0, stream>>>(Ows, Wo, bo, (float*)d_out);
}
